// HAN_13365938225234
// MI455X (gfx1250) — matrix operand order repaired and checked by reading, not compiled, not run on hardware
//
#include <hip/hip_runtime.h>

// ---------------------------------------------------------------------------
// Types for CDNA5 WMMA (wave32)
// ---------------------------------------------------------------------------
typedef __attribute__((ext_vector_type(16))) _Float16 v16h;
typedef __attribute__((ext_vector_type(8)))  _Float16 v8h;
typedef __attribute__((ext_vector_type(8)))  float    v8f;

#define HID   128
#define NHEAD 8
#define HD    16
#define NEG_SLOPE 0.2f

// ---------------------------------------------------------------------------
// fill
// ---------------------------------------------------------------------------
__global__ void fill_kernel(float* __restrict__ p, float v, long long n) {
    long long i = (long long)blockIdx.x * blockDim.x + threadIdx.x;
    long long stride = (long long)gridDim.x * blockDim.x;
    for (; i < n; i += stride) p[i] = v;
}

// ---------------------------------------------------------------------------
// convert + transpose f32 W[K][N] -> f16 Wt[N][K]
// ---------------------------------------------------------------------------
__global__ void cvt_transpose_kernel(const float* __restrict__ W,
                                     _Float16* __restrict__ Wt,
                                     int K, int Nn) {
    int idx = blockIdx.x * blockDim.x + threadIdx.x;
    if (idx >= K * Nn) return;
    int k = idx / Nn, n = idx % Nn;
    Wt[(size_t)n * K + k] = (_Float16)W[idx];
}

// ---------------------------------------------------------------------------
// Fused WMMA GEMM:  H = X[Nrows][K] @ W[K][128]  (W given transposed f16)
//   mode 0: Hout[row][col] = acc + bias[col]
//   mode 1: colsum[col] += sum_rows tanhf(acc + bias[col])   (semantic attn)
//   reluA : apply relu to X when staging (post-aggregation relu of HAN)
// Block: 256 threads = 8 waves; each wave computes a 16x128 output strip.
// Software-pipelined: global loads of k-tile (ks+1) overlap WMMA of tile ks.
// All 8 B fragments are loaded before the WMMA chain so the ds_loads drain
// once and the 8 v_wmma issue back-to-back.
// ---------------------------------------------------------------------------
__global__ __launch_bounds__(256) void gemm_wmma_kernel(
    const float* __restrict__ X, int K,
    const _Float16* __restrict__ Wt,      // [128][K]
    const float* __restrict__ bias,       // [128]
    float* __restrict__ Hout,             // mode 0
    float* __restrict__ colsum,           // mode 1
    int Nrows, int mode, int reluA)
{
    __shared__ __attribute__((aligned(32))) _Float16 sA[128 * 32];
    __shared__ __attribute__((aligned(32))) _Float16 sB[128 * 32]; // [n][k]

    const int tid  = threadIdx.x;
    const int wave = tid >> 5;
    const int lane = tid & 31;
    const int rowBlock = blockIdx.x * 128;

    // staging geometry: thread t owns 16 contiguous elements of the 128x32 tile
    const int idx = tid * 16;            // 0..4095
    const int ar  = idx >> 5;            // tile row (A) / tile n (B)
    const int ac  = idx & 31;            // 0 or 16
    int gr = rowBlock + ar;
    if (gr >= Nrows) gr = Nrows - 1;     // clamp; outputs guarded later
    const float*    aRow = X  + (size_t)gr * K + ac;
    const _Float16* bRow = Wt + (size_t)ar * K + ac;

    v8f acc[8];
#pragma unroll
    for (int i = 0; i < 8; ++i)
        acc[i] = (v8f){0.f,0.f,0.f,0.f,0.f,0.f,0.f,0.f};

    const int nsteps = K >> 5;

    // preload first k-tile into registers
    float4 f0 = *(const float4*)(aRow + 0);
    float4 f1 = *(const float4*)(aRow + 4);
    float4 f2 = *(const float4*)(aRow + 8);
    float4 f3 = *(const float4*)(aRow + 12);
    uint4  bv = *(const uint4*)(bRow);

    for (int ks = 0; ks < nsteps; ++ks) {
        // ---- commit staged registers to LDS ----
        {
            float tmp[16] = {f0.x,f0.y,f0.z,f0.w, f1.x,f1.y,f1.z,f1.w,
                             f2.x,f2.y,f2.z,f2.w, f3.x,f3.y,f3.z,f3.w};
            _Float16* d = sA + idx;
#pragma unroll
            for (int i = 0; i < 16; ++i) {
                float v = tmp[i];
                if (reluA) v = v > 0.f ? v : 0.f;
                d[i] = (_Float16)v;
            }
            *(uint4*)(sB + idx) = bv;
        }
        __syncthreads();

        // ---- issue next k-tile global loads (overlap with WMMA below) ----
        if (ks + 1 < nsteps) {
            const float*    na = aRow + (ks + 1) * 32;
            const _Float16* nb = bRow + (ks + 1) * 32;
            f0 = *(const float4*)(na + 0);
            f1 = *(const float4*)(na + 4);
            f2 = *(const float4*)(na + 8);
            f3 = *(const float4*)(na + 12);
            bv = *(const uint4*)(nb);
            if (ks + 2 < nsteps)
                __builtin_prefetch(aRow + (ks + 2) * 32, 0, 3);
        }

        // ---- fragments ----
        const int mrow  = wave * 16 + (lane & 15);
        const int abase = (lane < 16) ? 0 : 8;
        v8h alo = *(const v8h*)(sA + mrow * 32 + abase);       // K = abase..abase+7
        v8h ahi = *(const v8h*)(sA + mrow * 32 + 16 + abase);  // K = 16+abase..
        v16h afrag = __builtin_shufflevector(alo, ahi,
                       0,1,2,3,4,5,6,7,8,9,10,11,12,13,14,15);
        const int kg = (lane < 16) ? 0 : 8;

        v16h barr[8];
#pragma unroll
        for (int nt = 0; nt < 8; ++nt) {
            int n = nt * 16 + (lane & 15);
            barr[nt] = __builtin_shufflevector(*(const v8h*)(sB + n * 32 + kg),
                           *(const v8h*)(sB + n * 32 + 16 + kg),
                           0,1,2,3,4,5,6,7,8,9,10,11,12,13,14,15);
        }
#pragma unroll
        for (int nt = 0; nt < 8; ++nt) {
            acc[nt] = __builtin_amdgcn_wmma_f32_16x16x32_f16(
                          false, afrag, false, barr[nt],
                          (short)0, acc[nt], false, false);
        }
        __syncthreads();
    }

    // ---- epilogue ----
    const int mbase = rowBlock + wave * 16 + ((lane < 16) ? 0 : 8);
    const int col0  = lane & 15;
    if (mode == 0) {
#pragma unroll
        for (int nt = 0; nt < 8; ++nt) {
            int col = nt * 16 + col0;
            float b = bias[col];
#pragma unroll
            for (int v = 0; v < 8; ++v) {
                int row = mbase + v;
                if (row < Nrows)
                    Hout[(size_t)row * HID + col] = acc[nt][v] + b;
            }
        }
    } else {
#pragma unroll
        for (int nt = 0; nt < 8; ++nt) {
            int col = nt * 16 + col0;
            float b = bias[col];
            float local = 0.f;
#pragma unroll
            for (int v = 0; v < 8; ++v) {
                int row = mbase + v;
                if (row < Nrows) local += tanhf(acc[nt][v] + b);
            }
            atomicAdd(&colsum[col], local);
        }
    }
}

// ---------------------------------------------------------------------------
// per-(node,head) attention score: a[n][h] = sum_d h[n][h][d] * att[h][d]
// ---------------------------------------------------------------------------
__global__ void scores_kernel(const float* __restrict__ Hfeat,
                              const float* __restrict__ att,
                              float* __restrict__ out, int N) {
    int idx = blockIdx.x * blockDim.x + threadIdx.x;
    if (idx >= N * NHEAD) return;
    int n = idx >> 3, h = idx & 7;
    const float* hp = Hfeat + (size_t)n * HID + h * HD;
    const float* ap = att + h * HD;
    float s = 0.f;
#pragma unroll
    for (int d = 0; d < HD; d += 4) {
        float4 x = *(const float4*)(hp + d);
        float4 a = *(const float4*)(ap + d);
        s += x.x*a.x + x.y*a.y + x.z*a.z + x.w*a.w;
    }
    out[idx] = s;
}

// monotone-float atomic max (int max for >=0, uint min for <0)
__device__ __forceinline__ void atomicMaxF(float* addr, float val) {
    if (val >= 0.f) atomicMax((int*)addr, __float_as_int(val));
    else            atomicMin((unsigned int*)addr, __float_as_uint(val));
}

__device__ __forceinline__ float leaky(float a) {
    return a > 0.f ? a : NEG_SLOPE * a;
}

// ---------------------------------------------------------------------------
// edge pass 1: segment max of leaky_relu(alpha) over dst
// ---------------------------------------------------------------------------
__global__ void edge_max_kernel(const int* __restrict__ ei, int E,
                                const float* __restrict__ as_,
                                const float* __restrict__ ad_,
                                float* __restrict__ m) {
    int e = blockIdx.x * blockDim.x + threadIdx.x;
    if (e >= E) return;
    int src = ei[e], dst = ei[E + e];
#pragma unroll
    for (int h = 0; h < NHEAD; ++h) {
        float a = leaky(as_[src * NHEAD + h] + ad_[dst * NHEAD + h]);
        atomicMaxF(&m[dst * NHEAD + h], a);
    }
}

// ---------------------------------------------------------------------------
// edge pass 2: segment sum of exp(alpha - m[dst])
// ---------------------------------------------------------------------------
__global__ void edge_sum_kernel(const int* __restrict__ ei, int E,
                                const float* __restrict__ as_,
                                const float* __restrict__ ad_,
                                const float* __restrict__ m,
                                float* __restrict__ s) {
    int e = blockIdx.x * blockDim.x + threadIdx.x;
    if (e >= E) return;
    int src = ei[e], dst = ei[E + e];
#pragma unroll
    for (int h = 0; h < NHEAD; ++h) {
        float a = leaky(as_[src * NHEAD + h] + ad_[dst * NHEAD + h]);
        atomicAdd(&s[dst * NHEAD + h], expf(a - m[dst * NHEAD + h]));
    }
}

// ---------------------------------------------------------------------------
// edge pass 3: out[dst] += xs[src] * softmax_weight   (thread = edge x 4 comps)
// ---------------------------------------------------------------------------
__global__ void edge_msg_kernel(const int* __restrict__ ei, int E,
                                const float* __restrict__ as_,
                                const float* __restrict__ ad_,
                                const float* __restrict__ m,
                                const float* __restrict__ s,
                                const float* __restrict__ Xs,
                                float* __restrict__ Out) {
    long long idx = (long long)blockIdx.x * blockDim.x + threadIdx.x;
    if (idx >= (long long)E * 32) return;
    int e    = (int)(idx >> 5);
    int lane = (int)(idx & 31);
    int c    = lane * 4;          // component base 0..124
    int h    = c >> 4;            // head
    int src = ei[e], dst = ei[E + e];
    float a = leaky(as_[src * NHEAD + h] + ad_[dst * NHEAD + h]);
    float w = expf(a - m[dst * NHEAD + h]) / (s[dst * NHEAD + h] + 1e-16f);
    float4 x = *(const float4*)(Xs + (size_t)src * HID + c);
    float* o = Out + (size_t)dst * HID + c;
    atomicAdd(o + 0, x.x * w);
    atomicAdd(o + 1, x.y * w);
    atomicAdd(o + 2, x.z * w);
    atomicAdd(o + 3, x.w * w);
}

// ---------------------------------------------------------------------------
// semantic softmax over M=2 metapaths
// ---------------------------------------------------------------------------
__global__ void semantic_softmax_kernel(const float* __restrict__ colsum,
                                        const float* __restrict__ q,
                                        float* __restrict__ attn, float invN) {
    __shared__ float r0[128], r1[128];
    int t = threadIdx.x;
    float qv = q[t];
    r0[t] = qv * colsum[t];
    r1[t] = qv * colsum[128 + t];
    __syncthreads();
    for (int off = 64; off > 0; off >>= 1) {
        if (t < off) { r0[t] += r0[t + off]; r1[t] += r1[t + off]; }
        __syncthreads();
    }
    if (t == 0) {
        float s0 = r0[0] * invN, s1 = r1[0] * invN;
        float mx = fmaxf(s0, s1);
        float e0 = expf(s0 - mx), e1 = expf(s1 - mx);
        float inv = 1.f / (e0 + e1);
        attn[0] = e0 * inv;
        attn[1] = e1 * inv;
    }
}

// ---------------------------------------------------------------------------
// final head: out = (attn0*relu(ow) + attn1*relu(oc)) @ Wout + bout
// block = 16 rows x 16 outs, Wout and combined rows staged in LDS
// ---------------------------------------------------------------------------
__global__ __launch_bounds__(256) void final_head_kernel(
    const float* __restrict__ ow, const float* __restrict__ oc,
    const float* __restrict__ attn,
    const float* __restrict__ Wout, const float* __restrict__ bout,
    float* __restrict__ out, int N) {
    __shared__ float sW[HID * 16];
    __shared__ float sR[16 * HID];
    int t = threadIdx.x;
    for (int i = t; i < HID * 16; i += 256) sW[i] = Wout[i];
    float a0 = attn[0], a1 = attn[1];
    int row0 = blockIdx.x * 16;
    for (int i = t; i < 16 * HID; i += 256) {
        int r = i >> 7, c = i & 127;
        int gr = row0 + r;
        float v = 0.f;
        if (gr < N) {
            float vw = ow[(size_t)gr * HID + c]; vw = vw > 0.f ? vw : 0.f;
            float vc = oc[(size_t)gr * HID + c]; vc = vc > 0.f ? vc : 0.f;
            v = a0 * vw + a1 * vc;
        }
        sR[i] = v;
    }
    __syncthreads();
    int r = t >> 4, o = t & 15;
    int gr = row0 + r;
    if (gr < N) {
        float acc = bout[o];
#pragma unroll 8
        for (int j = 0; j < HID; ++j) acc += sR[r * HID + j] * sW[j * 16 + o];
        out[(size_t)gr * 16 + o] = acc;
    }
}

// ---------------------------------------------------------------------------
// launcher
// ---------------------------------------------------------------------------
extern "C" void kernel_launch(void* const* d_in, const int* in_sizes, int n_in,
                              void* d_out, int out_size, void* d_ws, size_t ws_size,
                              hipStream_t stream) {
    const float* x_paper   = (const float*)d_in[0];
    const float* x_author  = (const float*)d_in[1];
    const float* Wp_paper  = (const float*)d_in[2];
    const float* bp_paper  = (const float*)d_in[3];
    const float* Wp_author = (const float*)d_in[4];
    const float* bp_author = (const float*)d_in[5];
    const float* att_src_w = (const float*)d_in[6];
    const float* att_dst_w = (const float*)d_in[7];
    const float* att_src_c = (const float*)d_in[8];
    const float* att_dst_c = (const float*)d_in[9];
    const float* kW        = (const float*)d_in[12];
    const float* kb        = (const float*)d_in[13];
    const float* q         = (const float*)d_in[14];
    const float* Wout      = (const float*)d_in[15];
    const float* bout      = (const float*)d_in[16];
    const int*   ei_writes = (const int*)d_in[17];
    const int*   ei_cites  = (const int*)d_in[18];

    const int IN = 256;
    const int NP = in_sizes[0] / IN;     // papers
    const int NA = in_sizes[1] / IN;     // authors
    const int E  = in_sizes[17] / 2;
    const int NMAX = NP > NA ? NP : NA;

    // ---- workspace carve-out (256B aligned) ----
    char* base = (char*)d_ws;
    size_t off = 0;
    auto alloc = [&](size_t bytes) -> void* {
        void* p = base + off;
        off = (off + bytes + 255) & ~(size_t)255;
        return p;
    };
    float* hp     = (float*)alloc((size_t)NP * HID * 4);
    float* ha     = (float*)alloc((size_t)NA * HID * 4);
    float* ow     = (float*)alloc((size_t)NP * HID * 4);
    float* oc     = (float*)alloc((size_t)NP * HID * 4);
    float* a_s    = (float*)alloc((size_t)NMAX * NHEAD * 4);
    float* a_d    = (float*)alloc((size_t)NMAX * NHEAD * 4);
    float* mbuf   = (float*)alloc((size_t)NP * NHEAD * 4);
    float* sbuf   = (float*)alloc((size_t)NP * NHEAD * 4);
    float* colsum = (float*)alloc(2 * HID * 4);
    float* attn   = (float*)alloc(2 * 4);
    _Float16* WtP = (_Float16*)alloc((size_t)HID * IN * 2);
    _Float16* WtA = (_Float16*)alloc((size_t)HID * IN * 2);
    _Float16* WtK = (_Float16*)alloc((size_t)HID * HID * 2);
    (void)ws_size; (void)n_in; (void)out_size;

    // ---- weight conversion (transposed f16) ----
    cvt_transpose_kernel<<<(IN * HID + 255) / 256, 256, 0, stream>>>(Wp_paper,  WtP, IN,  HID);
    cvt_transpose_kernel<<<(IN * HID + 255) / 256, 256, 0, stream>>>(Wp_author, WtA, IN,  HID);
    cvt_transpose_kernel<<<(HID * HID + 255) / 256, 256, 0, stream>>>(kW,       WtK, HID, HID);

    // ---- projections (WMMA) ----
    int gbP = (NP + 127) / 128, gbA = (NA + 127) / 128;
    gemm_wmma_kernel<<<gbP, 256, 0, stream>>>(x_paper,  IN, WtP, bp_paper,  hp, nullptr, NP, 0, 0);
    gemm_wmma_kernel<<<gbA, 256, 0, stream>>>(x_author, IN, WtA, bp_author, ha, nullptr, NA, 0, 0);

    const int EB = (E + 255) / 256;
    const long long EM = (long long)E * 32;
    const int EMB = (int)((EM + 255) / 256);

    // ---- edge type "writes": author -> paper ----
    scores_kernel<<<(NA * NHEAD + 255) / 256, 256, 0, stream>>>(ha, att_src_w, a_s, NA);
    scores_kernel<<<(NP * NHEAD + 255) / 256, 256, 0, stream>>>(hp, att_dst_w, a_d, NP);
    fill_kernel<<<2048, 256, 0, stream>>>(mbuf, -1e30f, (long long)NP * NHEAD);
    fill_kernel<<<2048, 256, 0, stream>>>(sbuf, 0.f,    (long long)NP * NHEAD);
    fill_kernel<<<4096, 256, 0, stream>>>(ow,   0.f,    (long long)NP * HID);
    edge_max_kernel<<<EB, 256, 0, stream>>>(ei_writes, E, a_s, a_d, mbuf);
    edge_sum_kernel<<<EB, 256, 0, stream>>>(ei_writes, E, a_s, a_d, mbuf, sbuf);
    edge_msg_kernel<<<EMB, 256, 0, stream>>>(ei_writes, E, a_s, a_d, mbuf, sbuf, ha, ow);

    // ---- edge type "cites": paper -> paper ----
    scores_kernel<<<(NP * NHEAD + 255) / 256, 256, 0, stream>>>(hp, att_src_c, a_s, NP);
    scores_kernel<<<(NP * NHEAD + 255) / 256, 256, 0, stream>>>(hp, att_dst_c, a_d, NP);
    fill_kernel<<<2048, 256, 0, stream>>>(mbuf, -1e30f, (long long)NP * NHEAD);
    fill_kernel<<<2048, 256, 0, stream>>>(sbuf, 0.f,    (long long)NP * NHEAD);
    fill_kernel<<<4096, 256, 0, stream>>>(oc,   0.f,    (long long)NP * HID);
    edge_max_kernel<<<EB, 256, 0, stream>>>(ei_cites, E, a_s, a_d, mbuf);
    edge_sum_kernel<<<EB, 256, 0, stream>>>(ei_cites, E, a_s, a_d, mbuf, sbuf);
    edge_msg_kernel<<<EMB, 256, 0, stream>>>(ei_cites, E, a_s, a_d, mbuf, sbuf, hp, oc);

    // ---- semantic attention (WMMA tanh column-sum reduction) ----
    fill_kernel<<<1, 256, 0, stream>>>(colsum, 0.f, 2 * HID);
    gemm_wmma_kernel<<<gbP, 256, 0, stream>>>(ow, HID, WtK, kb, nullptr, colsum,       NP, 1, 1);
    gemm_wmma_kernel<<<gbP, 256, 0, stream>>>(oc, HID, WtK, kb, nullptr, colsum + HID, NP, 1, 1);
    semantic_softmax_kernel<<<1, 128, 0, stream>>>(colsum, q, attn, 1.f / (float)NP);

    // ---- final head ----
    final_head_kernel<<<(NP + 15) / 16, 256, 0, stream>>>(ow, oc, attn, Wout, bout,
                                                          (float*)d_out, NP);
}
